// MACTransformerBlock_66073776881782
// MI455X (gfx1250) — matrix operand order repaired and checked by reading, not compiled, not run on hardware
//
#include <hip/hip_runtime.h>
#include <hip/hip_bf16.h>

typedef __attribute__((ext_vector_type(16))) _Float16 v16h;
typedef __attribute__((ext_vector_type(8)))  float    v8f;
typedef __attribute__((ext_vector_type(4)))  float    f32x4;

union Frag16 { v16h h; f32x4 f[2]; };

#define BATCH   2
#define TSEG    1568
#define DM      768
#define NTOK    3140
#define NP2     3168      // padded token count (multiple of 32)
#define NHEADS  12
#define DHEAD   64
#define DFF     3072
#define OBROWS  1584      // attention output rows per batch: 1568..3151 -> local 0..1583
#define QTILES  99        // query tiles of 16 covering rows 1568..3151
#define KTILES  99        // key tiles of 32 covering 0..3167

__device__ __forceinline__ int tok_time(int i) {
  if (i < 4)    return 0;
  if (i < 1572) return 1 + (i - 4) / 196;
  if (i < NTOK) return 1 + (i - 1572) / 196;
  return 1000; // padded tokens: always masked
}

__device__ __forceinline__ v8f zero8() {
  v8f z;
#pragma unroll
  for (int i = 0; i < 8; i++) z[i] = 0.f;
  return z;
}

// ---------------------------------------------------------------------------
// Generic WMMA GEMM:  out[b,m,n] = epilogue( sum_k A[b,m,k] * Bw[n,k] )
// Bw row-major [Nn, K]  (i.e. computes A @ Bw^T).  Wave tile: 32(M) x 64(N).
// flags: 1 = f16 output, 2 = SiLU, 4 = add residual (f32)
// ---------------------------------------------------------------------------
__global__ __launch_bounds__(256)
void gemm_wmma(const _Float16* __restrict__ A, long long strideA, int lda,
               const _Float16* __restrict__ Bw, int ldb,
               const float* __restrict__ bias,
               const float* __restrict__ resid, long long strideRes,
               void* __restrict__ out, long long strideOut, int ldo,
               int Mb, int Nn, int K, int tilesM, int tilesN, int nb, int flags) {
  int w = blockIdx.x * 8 + (threadIdx.x >> 5);
  if (w >= nb * tilesM * tilesN) return;
  int tn = w % tilesN;
  int tm = (w / tilesN) % tilesM;
  int b  = w / (tilesN * tilesM);
  int lane = threadIdx.x & 31;
  int half = lane >> 4;
  int ln   = lane & 15;
  int m0 = tm * 32, n0 = tn * 64;
  const _Float16* Ab = A + (long long)b * strideA;

  v8f acc[2][4];
#pragma unroll
  for (int mt = 0; mt < 2; mt++)
#pragma unroll
    for (int nt = 0; nt < 4; nt++) acc[mt][nt] = zero8();

  int apat = half ? 8 : 0;    // A fragment k-offset pattern
  int bpat = half ? 8 : 0;    // B fragment k-offset pattern
  int rA0 = min(m0 + ln,      Mb - 1);
  int rA1 = min(m0 + 16 + ln, Mb - 1);

  for (int k = 0; k < K; k += 32) {
    Frag16 a0, a1;
    const _Float16* pa0 = Ab + (size_t)rA0 * lda + k + apat;
    const _Float16* pa1 = Ab + (size_t)rA1 * lda + k + apat;
    a0.f[0] = *(const f32x4*)pa0; a0.f[1] = *(const f32x4*)(pa0 + 16);
    a1.f[0] = *(const f32x4*)pa1; a1.f[1] = *(const f32x4*)(pa1 + 16);
#pragma unroll
    for (int nt = 0; nt < 4; nt++) {
      Frag16 bf;
      const _Float16* pb = Bw + (size_t)(n0 + nt * 16 + ln) * ldb + k + bpat;
      bf.f[0] = *(const f32x4*)pb; bf.f[1] = *(const f32x4*)(pb + 16);
      acc[0][nt] = __builtin_amdgcn_wmma_f32_16x16x32_f16(false, a0.h, false, bf.h,
                                                          (short)0, acc[0][nt], false, false);
      acc[1][nt] = __builtin_amdgcn_wmma_f32_16x16x32_f16(false, a1.h, false, bf.h,
                                                          (short)0, acc[1][nt], false, false);
    }
  }

#pragma unroll
  for (int mt = 0; mt < 2; mt++)
#pragma unroll
    for (int nt = 0; nt < 4; nt++)
#pragma unroll
      for (int r = 0; r < 8; r++) {
        int m = m0 + mt * 16 + r + 8 * half;
        if (m >= Mb) continue;
        int n = n0 + nt * 16 + ln;
        float v = acc[mt][nt][r];
        if (bias) v += bias[n];
        if (flags & 2) v = v / (1.f + __expf(-v));            // SiLU
        if (flags & 4) v += resid[(long long)b * strideRes + (size_t)m * ldo + n];
        long long oi = (long long)b * strideOut + (size_t)m * ldo + n;
        if (flags & 1) ((_Float16*)out)[oi] = (_Float16)v;
        else           ((float*)out)[oi]    = v;
      }
}

// ---------------------------------------------------------------------------
// Flash attention with WMMA (QK^T and PV), analytic MAC mask, wave32, 1 wave/blk
// ---------------------------------------------------------------------------
__global__ __launch_bounds__(32)
void attn_kernel(const _Float16* __restrict__ Qh, const _Float16* __restrict__ Kh,
                 const _Float16* __restrict__ Vt, _Float16* __restrict__ Ob) {
  __shared__ _Float16 ldsP[16 * 32];
  int w  = blockIdx.x;
  int qt = w % QTILES;
  int h  = (w / QTILES) % NHEADS;
  int b  = w / (QTILES * NHEADS);
  int lane = threadIdx.x;
  int half = lane >> 4;
  int ln   = lane & 15;
  int apat = half ? 8 : 0;
  int bpat = half ? 8 : 0;

  size_t bh = (size_t)b * NHEADS + h;
  const _Float16* Qp = Qh + bh * (size_t)NP2 * DHEAD;
  const _Float16* Kp = Kh + bh * (size_t)NP2 * DHEAD;
  const _Float16* Vp = Vt + bh * (size_t)DHEAD * NP2;
  int qbase = 1568 + qt * 16;

  // Q A-fragments (two k-steps over dh=64), kept in registers
  Frag16 qa[2];
#pragma unroll
  for (int ks = 0; ks < 2; ks++) {
    const _Float16* p = Qp + (size_t)(qbase + ln) * DHEAD + ks * 32 + apat;
    qa[ks].f[0] = *(const f32x4*)p;
    qa[ks].f[1] = *(const f32x4*)(p + 16);
  }

  int ti8[8];
#pragma unroll
  for (int r = 0; r < 8; r++) ti8[r] = tok_time(qbase + r + 8 * half);
  int tq = tok_time(min(qbase + ln, NTOK - 1));
#pragma unroll
  for (int s = 1; s < 16; s <<= 1) tq = max(tq, __shfl_xor(tq, s, 16));

  float mrow[8], lrow[8];
  v8f o[4];
#pragma unroll
  for (int r = 0; r < 8; r++) { mrow[r] = -1e30f; lrow[r] = 0.f; }
#pragma unroll
  for (int nf = 0; nf < 4; nf++) o[nf] = zero8();

  for (int kt = 0; kt < KTILES; kt++) {
    int kb = kt * 32;
    int tmin = (kb < 1572 && kb + 31 >= 1572) ? 1 : tok_time(kb);
    if (tmin > tq) continue; // key tile fully in the future -> fully masked

    // S = Q K^T (two 16-key sub-tiles, two k-steps each)
    v8f s[2];
#pragma unroll
    for (int sub = 0; sub < 2; sub++) {
      v8f c = zero8();
#pragma unroll
      for (int ks = 0; ks < 2; ks++) {
        Frag16 kf;
        const _Float16* p = Kp + (size_t)(kb + sub * 16 + ln) * DHEAD + ks * 32 + bpat;
        kf.f[0] = *(const f32x4*)p;
        kf.f[1] = *(const f32x4*)(p + 16);
        c = __builtin_amdgcn_wmma_f32_16x16x32_f16(false, qa[ks].h, false, kf.h,
                                                   (short)0, c, false, false);
      }
      s[sub] = c;
    }

    int j0 = kb + ln, j1 = kb + 16 + ln;
    int tj0 = tok_time(j0), tj1 = tok_time(j1);
    bool vv0 = j0 < NTOK, vv1 = j1 < NTOK;

    float pr0[8], pr1[8];
#pragma unroll
    for (int r = 0; r < 8; r++) {
      bool ok0 = vv0 && (tj0 <= ti8[r]);
      bool ok1 = vv1 && (tj1 <= ti8[r]);
      float s0 = ok0 ? s[0][r] * 0.125f : -1e30f;
      float s1 = ok1 ? s[1][r] * 0.125f : -1e30f;
      float mx = fmaxf(s0, s1);
#pragma unroll
      for (int sh = 1; sh < 16; sh <<= 1) mx = fmaxf(mx, __shfl_xor(mx, sh, 16));
      float mnew = fmaxf(mrow[r], mx);
      float fsc  = __expf(mrow[r] - mnew);
      float p0 = ok0 ? __expf(s0 - mnew) : 0.f;
      float p1 = ok1 ? __expf(s1 - mnew) : 0.f;
      float rs = p0 + p1;
#pragma unroll
      for (int sh = 1; sh < 16; sh <<= 1) rs += __shfl_xor(rs, sh, 16);
      lrow[r] = lrow[r] * fsc + rs;
      mrow[r] = mnew;
#pragma unroll
      for (int nf = 0; nf < 4; nf++) o[nf][r] *= fsc;
      pr0[r] = p0; pr1[r] = p1;
    }

    // C-layout -> A-layout for P via LDS
    __syncthreads();
#pragma unroll
    for (int r = 0; r < 8; r++) {
      int m = r + 8 * half;
      ldsP[m * 32 + ln]      = (_Float16)pr0[r];
      ldsP[m * 32 + 16 + ln] = (_Float16)pr1[r];
    }
    __syncthreads();
    Frag16 pa;
    {
      const _Float16* lp = &ldsP[ln * 32 + apat];
      pa.f[0] = *(const f32x4*)lp;
      pa.f[1] = *(const f32x4*)(lp + 16);
    }
    // O += P @ V  (V stored transposed: Vt[d, key])
#pragma unroll
    for (int nf = 0; nf < 4; nf++) {
      Frag16 vf;
      const _Float16* p = Vp + (size_t)(nf * 16 + ln) * NP2 + kb + bpat;
      vf.f[0] = *(const f32x4*)p;
      vf.f[1] = *(const f32x4*)(p + 16);
      o[nf] = __builtin_amdgcn_wmma_f32_16x16x32_f16(false, pa.h, false, vf.h,
                                                     (short)0, o[nf], false, false);
    }
  }

  // normalize + store (Ob: [B, OBROWS, 768], heads interleaved in D)
#pragma unroll
  for (int r = 0; r < 8; r++) {
    float inv = lrow[r] > 0.f ? 1.f / lrow[r] : 0.f;
    int rowL = qt * 16 + r + 8 * half; // local row = global row - 1568
    size_t base = ((size_t)b * OBROWS + rowL) * DM + h * DHEAD + ln;
#pragma unroll
    for (int nf = 0; nf < 4; nf++)
      Ob[base + nf * 16] = (_Float16)(o[nf][r] * inv);
  }
}

// ---------------------------------------------------------------------------
// Helpers: conversions, LN, QKV split
// ---------------------------------------------------------------------------
__global__ __launch_bounds__(256)
void cvt_f16(const float* __restrict__ s, _Float16* __restrict__ d, int n) {
  int i = blockIdx.x * 256 + threadIdx.x;
  if (i < n) d[i] = (_Float16)s[i];
}

__global__ __launch_bounds__(256)
void transpose_cvt768(const float* __restrict__ s, _Float16* __restrict__ d) {
  int i = blockIdx.x * 256 + threadIdx.x;
  if (i < 768 * 768) {
    int rr = i / 768, cc = i % 768;
    d[i] = (_Float16)s[cc * 768 + rr]; // d[n,k] = M[k,n]
  }
}

__device__ __forceinline__ float block_sum(float v, float* red) {
  int t = threadIdx.x;
  red[t] = v; __syncthreads();
  for (int s = 128; s > 0; s >>= 1) { if (t < s) red[t] += red[t + s]; __syncthreads(); }
  float r = red[0]; __syncthreads();
  return r;
}

// concat [P | h | x_seg] -> LN1 -> xaug(f32) -> attn_LN -> xa(f16, padded rows zeroed)
__global__ __launch_bounds__(256)
void build_xaug_ln(const float* __restrict__ P, const float* __restrict__ Hbuf,
                   const float* __restrict__ xseg,
                   const float* __restrict__ g1, const float* __restrict__ b1v,
                   const float* __restrict__ ga, const float* __restrict__ ba,
                   float* __restrict__ xaug, _Float16* __restrict__ xa) {
  __shared__ float red[256];
  int blk = blockIdx.x;
  int b = blk / NP2, i = blk % NP2, t = threadIdx.x;
  _Float16* xar = xa + ((size_t)b * NP2 + i) * DM;
  if (i >= NTOK) {
    for (int c = t; c < DM; c += 256) xar[c] = (_Float16)0.f;
    return;
  }
  const float* src;
  if (i < 4)         src = P    + (size_t)i * DM;
  else if (i < 1572) src = Hbuf + ((size_t)b * TSEG + (i - 4))    * DM;
  else               src = xseg + ((size_t)b * TSEG + (i - 1572)) * DM;

  float x0 = src[t], x1 = src[t + 256], x2 = src[t + 512];
  float mu = block_sum(x0 + x1 + x2, red) * (1.f / 768.f);
  float d0 = x0 - mu, d1 = x1 - mu, d2 = x2 - mu;
  float var = block_sum(d0 * d0 + d1 * d1 + d2 * d2, red) * (1.f / 768.f);
  float rs = rsqrtf(var + 1e-5f);
  float y0 = d0 * rs * g1[t]       + b1v[t];
  float y1 = d1 * rs * g1[t + 256] + b1v[t + 256];
  float y2 = d2 * rs * g1[t + 512] + b1v[t + 512];
  float* xr = xaug + ((size_t)b * NTOK + i) * DM;
  xr[t] = y0; xr[t + 256] = y1; xr[t + 512] = y2;

  float mu2 = block_sum(y0 + y1 + y2, red) * (1.f / 768.f);
  float e0 = y0 - mu2, e1 = y1 - mu2, e2 = y2 - mu2;
  float var2 = block_sum(e0 * e0 + e1 * e1 + e2 * e2, red) * (1.f / 768.f);
  float rs2 = rsqrtf(var2 + 1e-5f);
  xar[t]       = (_Float16)(e0 * rs2 * ga[t]       + ba[t]);
  xar[t + 256] = (_Float16)(e1 * rs2 * ga[t + 256] + ba[t + 256]);
  xar[t + 512] = (_Float16)(e2 * rs2 * ga[t + 512] + ba[t + 512]);
}

__global__ __launch_bounds__(256)
void ln_rows(const float* __restrict__ in, const float* __restrict__ g,
             const float* __restrict__ bb, _Float16* __restrict__ outp) {
  __shared__ float red[256];
  int row = blockIdx.x, t = threadIdx.x;
  const float* src = in + (size_t)row * DM;
  float x0 = src[t], x1 = src[t + 256], x2 = src[t + 512];
  float mu = block_sum(x0 + x1 + x2, red) * (1.f / 768.f);
  float d0 = x0 - mu, d1 = x1 - mu, d2 = x2 - mu;
  float var = block_sum(d0 * d0 + d1 * d1 + d2 * d2, red) * (1.f / 768.f);
  float rs = rsqrtf(var + 1e-5f);
  _Float16* o = outp + (size_t)row * DM;
  o[t]       = (_Float16)(d0 * rs * g[t]       + bb[t]);
  o[t + 256] = (_Float16)(d1 * rs * g[t + 256] + bb[t + 256]);
  o[t + 512] = (_Float16)(d2 * rs * g[t + 512] + bb[t + 512]);
}

__global__ __launch_bounds__(256)
void split_qkv(const _Float16* __restrict__ QKV, _Float16* __restrict__ Q,
               _Float16* __restrict__ K, _Float16* __restrict__ V) {
  int idx = blockIdx.x * 256 + threadIdx.x;
  if (idx >= BATCH * NP2 * DM) return;
  int hd = idx % DM;
  int r  = (idx / DM) % NP2;
  int b  = idx / (DM * NP2);
  int h = hd >> 6, d = hd & 63;
  const _Float16* row = QKV + ((size_t)b * NP2 + r) * (3 * DM);
  size_t bh = (size_t)b * NHEADS + h;
  Q[(bh * NP2 + r) * DHEAD + d] = row[hd];
  K[(bh * NP2 + r) * DHEAD + d] = row[DM + hd];
  V[(bh * DHEAD + d) * NP2 + r] = row[2 * DM + hd]; // transposed: Vt[d, key]
}

// ---------------------------------------------------------------------------
extern "C" void kernel_launch(void* const* d_in, const int* in_sizes, int n_in,
                              void* d_out, int out_size, void* d_ws, size_t ws_size,
                              hipStream_t stream) {
  (void)in_sizes; (void)n_in; (void)out_size; (void)ws_size;
  const float* x_seg = (const float*)d_in[0];
  const float* P     = (const float*)d_in[1];
  const float* Wq_w  = (const float*)d_in[2];
  const float* Wq_b  = (const float*)d_in[3];
  const float* Mw    = (const float*)d_in[4];
  const float* ln1_g = (const float*)d_in[5];
  const float* ln1_b = (const float*)d_in[6];
  const float* aln_g = (const float*)d_in[7];
  const float* aln_b = (const float*)d_in[8];
  const float* Wqkv  = (const float*)d_in[9];
  const float* Wo_w  = (const float*)d_in[10];
  const float* Wo_b  = (const float*)d_in[11];
  const float* ln2_g = (const float*)d_in[12];
  const float* ln2_b = (const float*)d_in[13];
  const float* ff_w1 = (const float*)d_in[14];
  const float* ff_b1 = (const float*)d_in[15];
  const float* ff_w2 = (const float*)d_in[16];
  const float* ff_b2 = (const float*)d_in[17];
  // d_in[18] (mask) unused: mask computed analytically in attn_kernel

  const int Mrows = BATCH * TSEG; // 3136
  char* ws = (char*)d_ws;
  size_t off = 0;
  auto alloc = [&](size_t bytes) -> char* {
    char* p = ws + off;
    off = (off + bytes + 255) & ~(size_t)255;
    return p;
  };
  _Float16* Xh    = (_Float16*)alloc((size_t)Mrows * DM * 2);
  _Float16* WqH   = (_Float16*)alloc((size_t)DM * DM * 2);
  _Float16* MtH   = (_Float16*)alloc((size_t)DM * DM * 2);
  _Float16* WqkvH = (_Float16*)alloc((size_t)3 * DM * DM * 2);
  _Float16* WoH   = (_Float16*)alloc((size_t)DM * DM * 2);
  _Float16* FF1H  = (_Float16*)alloc((size_t)DFF * DM * 2);
  _Float16* FF2H  = (_Float16*)alloc((size_t)DM * DFF * 2);
  _Float16* QtH   = (_Float16*)alloc((size_t)Mrows * DM * 2);
  float*    Hbuf  = (float*)   alloc((size_t)Mrows * DM * 4);
  float*    xaug  = (float*)   alloc((size_t)BATCH * NTOK * DM * 4);
  _Float16* xaH   = (_Float16*)alloc((size_t)BATCH * NP2 * DM * 2);
  _Float16* QKVh  = (_Float16*)alloc((size_t)BATCH * NP2 * 3 * DM * 2);
  _Float16* Qh    = (_Float16*)alloc((size_t)BATCH * NHEADS * NP2 * DHEAD * 2);
  _Float16* Kh    = (_Float16*)alloc((size_t)BATCH * NHEADS * NP2 * DHEAD * 2);
  _Float16* Vt    = (_Float16*)alloc((size_t)BATCH * NHEADS * DHEAD * NP2 * 2);
  _Float16* Ob    = (_Float16*)alloc((size_t)BATCH * OBROWS * DM * 2);
  // buffer reuse (lifetimes disjoint): xpost<-Hbuf, y2H<-QtH, ffaH<-QKVh
  float*    xpost = Hbuf;
  _Float16* y2H   = QtH;
  _Float16* ffaH  = QKVh;

  auto gemm = [&](const _Float16* A, long long sA, int lda, const _Float16* Bw, int ldb,
                  const float* bias, const float* resid, long long sR,
                  void* out, long long sO, int ldo, int Mb, int Nn, int K, int nb, int flags) {
    int tM = (Mb + 31) / 32, tN = Nn / 64;
    int waves = nb * tM * tN;
    gemm_wmma<<<(waves + 7) / 8, 256, 0, stream>>>(A, sA, lda, Bw, ldb, bias, resid, sR,
                                                   out, sO, ldo, Mb, Nn, K, tM, tN, nb, flags);
  };

  int n;
  n = Mrows * DM;        cvt_f16<<<(n + 255) / 256, 256, 0, stream>>>(x_seg, Xh, n);
  n = DM * DM;           cvt_f16<<<(n + 255) / 256, 256, 0, stream>>>(Wq_w, WqH, n);
  n = DM * DM;           transpose_cvt768<<<(n + 255) / 256, 256, 0, stream>>>(Mw, MtH);
  n = 3 * DM * DM;       cvt_f16<<<(n + 255) / 256, 256, 0, stream>>>(Wqkv, WqkvH, n);
  n = DM * DM;           cvt_f16<<<(n + 255) / 256, 256, 0, stream>>>(Wo_w, WoH, n);
  n = DFF * DM;          cvt_f16<<<(n + 255) / 256, 256, 0, stream>>>(ff_w1, FF1H, n);
  n = DM * DFF;          cvt_f16<<<(n + 255) / 256, 256, 0, stream>>>(ff_w2, FF2H, n);

  // q_t = x_seg @ Wq^T + b  (f16)
  gemm(Xh, 0, DM, WqH, DM, Wq_b, nullptr, 0, QtH, 0, DM, Mrows, DM, DM, 1, 1);
  // h = q_t @ M  (via M^T stored row-major)  (f32)
  gemm(QtH, 0, DM, MtH, DM, nullptr, nullptr, 0, Hbuf, 0, DM, Mrows, DM, DM, 1, 0);
  // concat + LN1 + attn-LN
  build_xaug_ln<<<BATCH * NP2, 256, 0, stream>>>(P, Hbuf, x_seg, ln1_g, ln1_b,
                                                 aln_g, aln_b, xaug, xaH);
  // qkv = xa @ Wqkv^T  (f16, no bias)
  gemm(xaH, 0, DM, WqkvH, DM, nullptr, nullptr, 0, QKVh, 0, 3 * DM,
       BATCH * NP2, 3 * DM, DM, 1, 1);
  n = BATCH * NP2 * DM;
  split_qkv<<<(n + 255) / 256, 256, 0, stream>>>(QKVh, Qh, Kh, Vt);
  // flash attention (queries restricted to output rows 1568..3151)
  attn_kernel<<<BATCH * NHEADS * QTILES, 32, 0, stream>>>(Qh, Kh, Vt, Ob);
  // xpost = xaug[seg rows] + o @ Wo^T + b   (batched, f32)
  gemm(Ob + 4 * DM, (long long)OBROWS * DM, DM, WoH, DM, Wo_b,
       xaug + 1572 * DM, (long long)NTOK * DM,
       xpost, (long long)TSEG * DM, DM, TSEG, DM, DM, BATCH, 4);
  // y2 = LN2(xpost)  (f16)
  ln_rows<<<Mrows, 256, 0, stream>>>(xpost, ln2_g, ln2_b, y2H);
  // ffa = SiLU(y2 @ W1^T + b1)  (f16)
  gemm(y2H, 0, DM, FF1H, DM, ff_b1, nullptr, 0, ffaH, 0, DFF, Mrows, DFF, DM, 1, 1 | 2);
  // out = xpost + ffa @ W2^T + b2  (f32, final)
  gemm(ffaH, 0, DFF, FF2H, DFF, ff_b2, xpost, 0, d_out, 0, DM, Mrows, DM, DFF, 1, 4);
}
